// TGATModel_24816321036922
// MI455X (gfx1250) — matrix operand order repaired and checked by reading, not compiled, not run on hardware
//
#include <hip/hip_runtime.h>
#include <math.h>

// ---------------- model constants (match reference) ----------------
#define LAYERS 2
#define IN_F   172
#define HID    128
#define NH     4
#define HD     32          // HID / NH
#define TD     100
#define NNODE  150000
#define NEDGE  500000
#define NBATCH 15000

typedef __attribute__((ext_vector_type(16))) _Float16 v16h;
typedef __attribute__((ext_vector_type(8)))  _Float16 v8h;
typedef __attribute__((ext_vector_type(8)))  float    v8f;
typedef long long i64;

// ---------------- B-fragment packing ----------------
// Packs W (Kreal x Ncols row-major, fp32) into WMMA-B fragment order, f16,
// zero-padded to Kpad. Consumption: frag (ntile, kc) for lane l, element e is
//   value of W[kc*32 + (e<8 ? (l>>4)*8 + e : 16 + (l>>4)*8 + (e-8))][ntile*16 + (l&15)]
__global__ void pack_b_kernel(const float* __restrict__ W, _Float16* __restrict__ Bpk,
                              int Kreal, int Kpad, int Ncols) {
  int idx = blockIdx.x * blockDim.x + threadIdx.x;
  int nkc = Kpad / 32;
  int total = (Ncols / 16) * nkc * 512;
  if (idx >= total) return;
  int elem = idx & 15;
  int lane = (idx >> 4) & 31;
  int rem  = idx >> 9;
  int kc   = rem % nkc;
  int nt   = rem / nkc;
  int n  = nt * 16 + (lane & 15);
  int kk = kc * 32 + ((elem < 8) ? ((lane >> 4) * 8 + elem) : (16 + (lane >> 4) * 8 + (elem - 8)));
  float val = (kk < Kreal) ? W[(i64)kk * Ncols + n] : 0.0f;
  Bpk[idx] = (_Float16)val;
}

// ---------------- generic GEMM: C = [relu](A[ids?] @ W + bias) ----------------
// blockDim = 256 (8 waves); wave w owns column tile w; block owns 16 rows.
// A fp32 or f16 row-major; B pre-packed f16 fragments; C fp32 or f16.
template <int KREAL, int KPAD, int NCOLS, bool AHALF, bool GATHER, bool RELU, bool OUTHALF>
__global__ void gemm_wmma_t(const float* __restrict__ Af, const _Float16* __restrict__ Ah,
                            const int* __restrict__ ids,
                            const _Float16* __restrict__ Bpk, const float* __restrict__ bias,
                            float* __restrict__ Cf, _Float16* __restrict__ Ch,
                            int M, int ldA) {
  constexpr int NKC = KPAD / 32;
  int wave = threadIdx.x >> 5;
  int lane = threadIdx.x & 31;
  if (wave * 16 >= NCOLS) return;            // wave-uniform exit, EXEC stays full
  int r  = lane & 15;
  int hh = lane >> 4;
  int mrow = blockIdx.x * 16 + r;
  int rowc = (mrow < M) ? mrow : (M - 1);    // clamp: loads never diverge
  i64 rowBase = GATHER ? (i64)ids[rowc] * ldA : (i64)rowc * ldA;
  int n = wave * 16 + r;
  const v16h* Bp = (const v16h*)Bpk;

  v8f acc = {};
#pragma unroll
  for (int kc = 0; kc < NKC; ++kc) {
    const int k0 = kc * 32;
    v16h a;
    if ((kc + 1) * 32 <= KREAL) {
      if (AHALF) {
        // f16 A: two contiguous 8-half runs -> 2x b128, no conversion
        const v8h* p0 = (const v8h*)(Ah + rowBase + k0 + hh * 8);
        const v8h* p1 = (const v8h*)(Ah + rowBase + k0 + 16 + hh * 8);
        v8h f0 = p0[0], f1 = p1[0];
#pragma unroll
        for (int t = 0; t < 8; ++t) { a[t] = f0[t]; a[8 + t] = f1[t]; }
      } else {
        // fp32 A: two contiguous 8-float runs -> 4x b128 + pk-cvt
        const float4* p0 = (const float4*)(Af + rowBase + k0 + hh * 8);
        const float4* p1 = (const float4*)(Af + rowBase + k0 + 16 + hh * 8);
        float4 f0 = p0[0], f1 = p0[1];
        float4 f2 = p1[0], f3 = p1[1];
        a[0]  = (_Float16)f0.x; a[1]  = (_Float16)f0.y;
        a[2]  = (_Float16)f0.z; a[3]  = (_Float16)f0.w;
        a[4]  = (_Float16)f1.x; a[5]  = (_Float16)f1.y;
        a[6]  = (_Float16)f1.z; a[7]  = (_Float16)f1.w;
        a[8]  = (_Float16)f2.x; a[9]  = (_Float16)f2.y;
        a[10] = (_Float16)f2.z; a[11] = (_Float16)f2.w;
        a[12] = (_Float16)f3.x; a[13] = (_Float16)f3.y;
        a[14] = (_Float16)f3.z; a[15] = (_Float16)f3.w;
      }
    } else {
      // boundary chunk (K pad): clamped unconditional load + select
#pragma unroll
      for (int j = 0; j < 8; ++j) {
#pragma unroll
        for (int p = 0; p < 2; ++p) {
          int kl = (j < 4) ? (hh * 8 + 2 * j + p) : (16 + hh * 8 + 2 * (j - 4) + p);
          int kk = k0 + kl;
          int kks = (kk < KREAL) ? kk : 0;
          float val = AHALF ? (float)Ah[rowBase + kks] : Af[rowBase + kks];
          a[2 * j + p] = (_Float16)((kk < KREAL) ? val : 0.0f);
        }
      }
    }
    v16h b = Bp[(i64)(wave * NKC + kc) * 32 + lane];
    acc = __builtin_amdgcn_wmma_f32_16x16x32_f16(false, a, false, b,
                                                 (short)0, acc, false, false);
  }
  float bv = bias[n];
#pragma unroll
  for (int j = 0; j < 8; ++j) {
    int m = blockIdx.x * 16 + j + 8 * hh;
    float val = acc[j] + bv;
    if (RELU) val = fmaxf(val, 0.0f);
    if (m < M) {
      if (OUTHALF) Ch[(i64)m * NCOLS + n] = (_Float16)val;
      else         Cf[(i64)m * NCOLS + n] = val;
    }
  }
}

// ---------------- time-encoding GEMM: ebuf = cos(dt*freq+phase) @ We ----------
__global__ void te_gemm_wmma(const float* __restrict__ dts, const float* __restrict__ freq,
                             const float* __restrict__ phase, const _Float16* __restrict__ Wepk,
                             _Float16* __restrict__ ebuf, int E) {
  __shared__ float s_freq[128];
  __shared__ float s_phase[128];
  if (threadIdx.x < 128) {
    int t = threadIdx.x;
    s_freq[t]  = (t < TD) ? freq[t]  : 0.0f;
    s_phase[t] = (t < TD) ? phase[t] : 0.0f;
  }
  __syncthreads();

  int wave = threadIdx.x >> 5;
  int lane = threadIdx.x & 31;
  int r  = lane & 15;
  int hh = lane >> 4;
  int erow = blockIdx.x * 16 + r;
  int ec = (erow < E) ? erow : (E - 1);
  float dt = dts[ec];
  int n = wave * 16 + r;
  const v16h* Bp = (const v16h*)Wepk;

  v8f acc = {};
#pragma unroll
  for (int kc = 0; kc < 4; ++kc) {           // K = 100 padded to 128
    const int k0 = kc * 32;
    v16h a;
#pragma unroll
    for (int j = 0; j < 8; ++j) {
#pragma unroll
      for (int p = 0; p < 2; ++p) {
        int kl = (j < 4) ? (hh * 8 + 2 * j + p) : (16 + hh * 8 + 2 * (j - 4) + p);
        int kk = k0 + kl;
        float c = __cosf(dt * s_freq[kk] + s_phase[kk]);
        a[2 * j + p] = (_Float16)((kk < TD) ? c : 0.0f);
      }
    }
    v16h b = Bp[(i64)(wave * 4 + kc) * 32 + lane];
    acc = __builtin_amdgcn_wmma_f32_16x16x32_f16(false, a, false, b,
                                                 (short)0, acc, false, false);
  }
#pragma unroll
  for (int j = 0; j < 8; ++j) {
    int m = blockIdx.x * 16 + j + 8 * hh;
    if (m < E) ebuf[(i64)m * HID + n] = (_Float16)acc[j];
  }
}

// ---------------- helpers ----------------
__device__ inline void atomicMaxFloat(float* addr, float val) {
  unsigned int* ua = (unsigned int*)addr;
  unsigned int old = *ua;
  while (__uint_as_float(old) < val) {
    unsigned int assumed = old;
    old = atomicCAS(ua, assumed, __float_as_uint(val));
    if (old == assumed) break;
  }
}

__global__ void fill_f32(float* __restrict__ p, i64 n, float v) {
  i64 idx = (i64)blockIdx.x * blockDim.x + threadIdx.x;
  if (idx < n) p[idx] = v;
}

// ---------------- edge stage (scatter softmax attention, f16 inputs) ---------
__global__ void edge_alpha_kernel(const _Float16* __restrict__ q, const _Float16* __restrict__ kb,
                                  const _Float16* __restrict__ eb, const int* __restrict__ src,
                                  const int* __restrict__ dst, float* __restrict__ alpha,
                                  float* __restrict__ mmax, int E) {
  int idx = blockIdx.x * blockDim.x + threadIdx.x;
  if (idx >= E * NH) return;
  int e = idx >> 2, head = idx & 3;
  int s = src[e], d = dst[e];
  const v8h* qp = (const v8h*)(q  + (i64)d * HID + head * HD);
  const v8h* kp = (const v8h*)(kb + (i64)s * HID + head * HD);
  const v8h* ep = (const v8h*)(eb + (i64)e * HID + head * HD);
  float acc = 0.0f;
#pragma unroll
  for (int j = 0; j < HD / 8; ++j) {
    v8h qv = qp[j], kv = kp[j], ev = ep[j];
#pragma unroll
    for (int t = 0; t < 8; ++t)
      acc += (float)qv[t] * ((float)kv[t] + (float)ev[t]);
  }
  acc *= 0.17677669529663687f;  // 1/sqrt(32)
  alpha[idx] = acc;
  atomicMaxFloat(&mmax[d * NH + head], acc);
}

__global__ void edge_exp_kernel(const float* __restrict__ alpha, const float* __restrict__ mmax,
                                const int* __restrict__ dst, float* __restrict__ aexp,
                                float* __restrict__ ssum, int E) {
  int idx = blockIdx.x * blockDim.x + threadIdx.x;
  if (idx >= E * NH) return;
  int e = idx >> 2, head = idx & 3;
  int d = dst[e];
  float a = __expf(alpha[idx] - mmax[d * NH + head]);
  aexp[idx] = a;
  atomicAdd(&ssum[d * NH + head], a);
}

__global__ void edge_agg_kernel(const float* __restrict__ aexp, const float* __restrict__ ssum,
                                const _Float16* __restrict__ vb, const _Float16* __restrict__ eb,
                                const int* __restrict__ src, const int* __restrict__ dst,
                                float* __restrict__ agg, int E) {
  int idx = blockIdx.x * blockDim.x + threadIdx.x;
  if (idx >= E * NH) return;
  int e = idx >> 2, head = idx & 3;
  int s = src[e], d = dst[e];
  float w = aexp[idx] / (ssum[d * NH + head] + 1e-16f);
  const v8h* vp = (const v8h*)(vb + (i64)s * HID + head * HD);
  const v8h* ep = (const v8h*)(eb + (i64)e * HID + head * HD);
  float* op = agg + (i64)d * HID + head * HD;
#pragma unroll
  for (int j = 0; j < HD / 8; ++j) {
    v8h vv = vp[j], ev = ep[j];
#pragma unroll
    for (int t = 0; t < 8; ++t)
      atomicAdd(&op[8 * j + t], w * ((float)vv[t] + (float)ev[t]));
  }
}

// ---------------- batch-norm ----------------
// blockDim == Ncols; each block reduces 64 rows into per-column partials.
__global__ void col_stats(const float* __restrict__ X, int M, int Ncols, int relu_in,
                          float* __restrict__ sum, float* __restrict__ sumsq) {
  int c = threadIdx.x;
  i64 row0 = (i64)blockIdx.x * 64;
  float s = 0.0f, ss = 0.0f;
  for (int rr = 0; rr < 64; ++rr) {
    i64 m = row0 + rr;
    if (m < M) {
      float t = X[m * Ncols + c];
      if (relu_in) t = fmaxf(t, 0.0f);
      s += t; ss += t * t;
    }
  }
  atomicAdd(&sum[c], s);
  atomicAdd(&sumsq[c], ss);
}

// Yh != nullptr -> f16 output, else fp32 to Yf.
__global__ void bn_apply(const float* __restrict__ X, const float* __restrict__ sum,
                         const float* __restrict__ sumsq, const float* __restrict__ g,
                         const float* __restrict__ b, float* __restrict__ Yf,
                         _Float16* __restrict__ Yh,
                         int M, int Ncols, int relu_in, int relu_out) {
  i64 idx = (i64)blockIdx.x * blockDim.x + threadIdx.x;
  if (idx >= (i64)M * Ncols) return;
  int c = (int)(idx % Ncols);
  float mean = sum[c] / (float)M;
  float var  = sumsq[c] / (float)M - mean * mean;
  float t = X[idx];
  if (relu_in) t = fmaxf(t, 0.0f);
  float y = g[c] * (t - mean) * rsqrtf(var + 1e-5f) + b[c];
  if (relu_out) y = fmaxf(y, 0.0f);
  if (Yh) Yh[idx] = (_Float16)y;
  else    Yf[idx] = y;
}

// ---------------- head final dot ----------------
__global__ void head_final(const float* __restrict__ z2, const float* __restrict__ W3,
                           const float* __restrict__ b3, float* __restrict__ out, int Bn) {
  int r = blockIdx.x * blockDim.x + threadIdx.x;
  if (r >= Bn) return;
  float acc = b3[0];
  const float4* zp = (const float4*)(z2 + (i64)r * 64);
#pragma unroll
  for (int j = 0; j < 16; ++j) {
    float4 zv = zp[j];
    acc += zv.x * W3[4 * j] + zv.y * W3[4 * j + 1]
         + zv.z * W3[4 * j + 2] + zv.w * W3[4 * j + 3];
  }
  out[r] = acc;
}

// ---------------- host side ----------------
extern "C" void kernel_launch(void* const* d_in, const int* in_sizes, int n_in,
                              void* d_out, int out_size, void* d_ws, size_t ws_size,
                              hipStream_t stream) {
  (void)in_sizes; (void)n_in; (void)out_size; (void)ws_size;
  const float* x_all   = (const float*)d_in[0];
  const int*   ids     = (const int*)  d_in[1];
  const int*   e_src   = (const int*)  d_in[2];
  const int*   e_dst   = (const int*)  d_in[3];
  const float* dts     = (const float*)d_in[4];
  const float* freq    = (const float*)d_in[6];
  const float* phase   = (const float*)d_in[7];
  const float* Wp      = (const float*)d_in[8];
  const float* bp      = (const float*)d_in[9];
  const float* Wq      = (const float*)d_in[10];
  const float* bq      = (const float*)d_in[11];
  const float* Wk      = (const float*)d_in[12];
  const float* bk      = (const float*)d_in[13];
  const float* Wv      = (const float*)d_in[14];
  const float* bv      = (const float*)d_in[15];
  const float* We      = (const float*)d_in[16];
  const float* Wskip   = (const float*)d_in[17];
  const float* bskip   = (const float*)d_in[18];
  const float* bn_g    = (const float*)d_in[19];
  const float* bn_b    = (const float*)d_in[20];
  const float* W1      = (const float*)d_in[21];
  const float* b1      = (const float*)d_in[22];
  const float* g1      = (const float*)d_in[23];
  const float* be1     = (const float*)d_in[24];
  const float* W2      = (const float*)d_in[25];
  const float* b2      = (const float*)d_in[26];
  const float* g2      = (const float*)d_in[27];
  const float* be2     = (const float*)d_in[28];
  const float* W3      = (const float*)d_in[29];
  const float* b3      = (const float*)d_in[30];

  const i64 NH_ = (i64)NNODE * HID;
  // fp32 region
  float* ws    = (float*)d_ws;
  float* agg   = ws;                          // NNODE*HID (skip init + scatter)
  float* alpha = agg  + NH_;                  // NEDGE*NH
  float* aexp  = alpha + (i64)NEDGE * NH;     // NEDGE*NH
  float* mmax  = aexp  + (i64)NEDGE * NH;     // NNODE*NH
  float* ssum  = mmax  + (i64)NNODE * NH;     // NNODE*NH
  float* z1    = ssum  + (i64)NNODE * NH;     // NBATCH*HID (head)
  float* z2    = z1    + (i64)NBATCH * HID;   // NBATCH*64  (head)
  float* stat  = z2    + (i64)NBATCH * 64;    // 2*HID
  // f16 region (32B-aligned by construction)
  _Float16* h16 = (_Float16*)(stat + 2 * HID); // NNODE*HID
  _Float16* q16 = h16 + NH_;
  _Float16* k16 = q16 + NH_;
  _Float16* v16 = k16 + NH_;
  _Float16* e16 = v16 + NH_;                   // NEDGE*HID
  // packed weights (f16)
  _Float16* pkWp = e16 + (i64)NEDGE * HID;     // 192*128
  _Float16* pkWq = pkWp + 192 * 128;           // 2 x 128*128
  _Float16* pkWk = pkWq + 2 * 128 * 128;
  _Float16* pkWv = pkWk + 2 * 128 * 128;
  _Float16* pkWs = pkWv + 2 * 128 * 128;
  _Float16* pkWe = pkWs + 2 * 128 * 128;       // 2 x 128*128 (K=100 padded)
  _Float16* pkW1 = pkWe + 2 * 128 * 128;       // 128*128
  _Float16* pkW2 = pkW1 + 128 * 128;           // 128*64

  const int mtN = (NNODE + 15) / 16;           // 9375
  const int mtE = (NEDGE + 15) / 16;           // 31250
  const int mtB = (NBATCH + 15) / 16;          // 938
  const int eTH = (NEDGE * NH + 255) / 256;

  // ---- pack all weights into WMMA-B fragment order (tiny, once per launch) ----
  pack_b_kernel<<<(192 * 128 + 255) / 256, 256, 0, stream>>>(Wp, pkWp, IN_F, 192, HID);
  for (int i = 0; i < LAYERS; ++i) {
    pack_b_kernel<<<(128 * 128 + 255) / 256, 256, 0, stream>>>(Wq + (i64)i * HID * HID, pkWq + (i64)i * 128 * 128, HID, HID, HID);
    pack_b_kernel<<<(128 * 128 + 255) / 256, 256, 0, stream>>>(Wk + (i64)i * HID * HID, pkWk + (i64)i * 128 * 128, HID, HID, HID);
    pack_b_kernel<<<(128 * 128 + 255) / 256, 256, 0, stream>>>(Wv + (i64)i * HID * HID, pkWv + (i64)i * 128 * 128, HID, HID, HID);
    pack_b_kernel<<<(128 * 128 + 255) / 256, 256, 0, stream>>>(Wskip + (i64)i * HID * HID, pkWs + (i64)i * 128 * 128, HID, HID, HID);
    pack_b_kernel<<<(128 * 128 + 255) / 256, 256, 0, stream>>>(We + (i64)i * TD * HID, pkWe + (i64)i * 128 * 128, TD, 128, HID);
  }
  pack_b_kernel<<<(128 * 128 + 255) / 256, 256, 0, stream>>>(W1, pkW1, HID, HID, HID);
  pack_b_kernel<<<(128 * 64 + 255) / 256, 256, 0, stream>>>(W2, pkW2, HID, HID, 64);

  // ---- h16 = relu(x_all[ids] @ Wp + bp) ----
  gemm_wmma_t<IN_F, 192, HID, false, true, true, true><<<mtN, 256, 0, stream>>>(
      x_all, nullptr, ids, pkWp, bp, nullptr, h16, NNODE, IN_F);

  for (int i = 0; i < LAYERS; ++i) {
    const int*   src_i = e_src + (i64)i * NEDGE;
    const int*   dst_i = e_dst + (i64)i * NEDGE;
    const float* dts_i = dts + (i64)i * NEDGE;

    gemm_wmma_t<HID, HID, HID, true, false, false, true><<<mtN, 256, 0, stream>>>(
        nullptr, h16, nullptr, pkWq + (i64)i * 128 * 128, bq + i * HID, nullptr, q16, NNODE, HID);
    gemm_wmma_t<HID, HID, HID, true, false, false, true><<<mtN, 256, 0, stream>>>(
        nullptr, h16, nullptr, pkWk + (i64)i * 128 * 128, bk + i * HID, nullptr, k16, NNODE, HID);
    gemm_wmma_t<HID, HID, HID, true, false, false, true><<<mtN, 256, 0, stream>>>(
        nullptr, h16, nullptr, pkWv + (i64)i * 128 * 128, bv + i * HID, nullptr, v16, NNODE, HID);
    gemm_wmma_t<HID, HID, HID, true, false, false, false><<<mtN, 256, 0, stream>>>(
        nullptr, h16, nullptr, pkWs + (i64)i * 128 * 128, bskip + i * HID, agg, nullptr, NNODE, HID);

    te_gemm_wmma<<<mtE, 256, 0, stream>>>(dts_i, freq, phase,
                                          pkWe + (i64)i * 128 * 128, e16, NEDGE);

    fill_f32<<<((i64)NNODE * NH + 255) / 256, 256, 0, stream>>>(mmax, (i64)NNODE * NH, -3.0e38f);
    fill_f32<<<((i64)NNODE * NH + 255) / 256, 256, 0, stream>>>(ssum, (i64)NNODE * NH, 0.0f);

    edge_alpha_kernel<<<eTH, 256, 0, stream>>>(q16, k16, e16, src_i, dst_i, alpha, mmax, NEDGE);
    edge_exp_kernel  <<<eTH, 256, 0, stream>>>(alpha, mmax, dst_i, aexp, ssum, NEDGE);
    edge_agg_kernel  <<<eTH, 256, 0, stream>>>(aexp, ssum, v16, e16, src_i, dst_i, agg, NEDGE);

    fill_f32<<<1, 256, 0, stream>>>(stat, 2 * HID, 0.0f);
    col_stats<<<(NNODE + 63) / 64, HID, 0, stream>>>(agg, NNODE, HID, 1, stat, stat + HID);
    bn_apply<<<((i64)NNODE * HID + 255) / 256, 256, 0, stream>>>(
        agg, stat, stat + HID, bn_g + i * HID, bn_b + i * HID, nullptr, h16, NNODE, HID, 1, 0);
  }

  // ---- head: z1 = relu(bn(h16[:B] @ W1 + b1)) ----
  gemm_wmma_t<HID, HID, HID, true, false, false, false><<<mtB, 256, 0, stream>>>(
      nullptr, h16, nullptr, pkW1, b1, z1, nullptr, NBATCH, HID);
  fill_f32<<<1, 256, 0, stream>>>(stat, 2 * HID, 0.0f);
  col_stats<<<(NBATCH + 63) / 64, HID, 0, stream>>>(z1, NBATCH, HID, 0, stat, stat + HID);
  bn_apply<<<((i64)NBATCH * HID + 255) / 256, 256, 0, stream>>>(
      z1, stat, stat + HID, g1, be1, z1, nullptr, NBATCH, HID, 0, 1);

  // ---- z2 = relu(bn(z1 @ W2 + b2)) ----
  gemm_wmma_t<HID, HID, 64, false, false, false, false><<<mtB, 256, 0, stream>>>(
      z1, nullptr, nullptr, pkW2, b2, z2, nullptr, NBATCH, HID);
  fill_f32<<<1, 256, 0, stream>>>(stat, 2 * HID, 0.0f);
  col_stats<<<(NBATCH + 63) / 64, 64, 0, stream>>>(z2, NBATCH, 64, 0, stat, stat + 64);
  bn_apply<<<((i64)NBATCH * 64 + 255) / 256, 256, 0, stream>>>(
      z2, stat, stat + 64, g2, be2, z2, nullptr, NBATCH, 64, 0, 1);

  head_final<<<(NBATCH + 255) / 256, 256, 0, stream>>>(z2, W3, b3, (float*)d_out, NBATCH);
}
